// Encoder_6665789243929
// MI455X (gfx1250) — matrix operand order repaired and checked by reading, not compiled, not run on hardware
//
#include <hip/hip_runtime.h>
#include <hip/hip_bf16.h>

// ---------------- problem constants ----------------
constexpr int B = 128, T = 1024, I = 128, H = 512;
constexpr int G4 = 4 * H;        // 2048 gate rows
constexpr int KD = I + H;        // 640 fused K
constexpr int NWG = H / 8;       // 64 workgroups, each owns 8 H-columns
constexpr int BLK = 256;         // 8 waves of 32

// ---------------- workspace layout (bytes) ----------------
constexpr size_t OFF_M   = 0;                               // B*T f32 boundary mask
constexpr size_t OFF_S0  = OFF_M  + (size_t)B * T * 4;      // B*T i32 inverse slot map
constexpr size_t OFF_BS  = OFF_S0 + (size_t)B * T * 4;      // 4H f32 bias sum
constexpr size_t OFF_SY  = OFF_BS + (size_t)G4 * 4;         // sync: cnt, gen
constexpr size_t OFF_H16 = OFF_SY + 256;                    // 2 * B*H f16 (ping-pong h)
constexpr size_t OFF_WC  = OFF_H16 + (size_t)2 * B * H * 2; // G4 x KD f16 fused weights
constexpr size_t OFF_X16 = OFF_WC + (size_t)G4 * KD * 2;    // T*B*I f16 (t-major x)
// total ~= 37.4 MB

typedef __attribute__((ext_vector_type(16))) _Float16 v16h;
typedef __attribute__((ext_vector_type(8)))  float    v8f;

union F16x16 { v16h h; unsigned u[8]; };

__device__ __forceinline__ float sigf(float x) { return 1.0f / (1.0f + __expf(-x)); }

// ---------------- prep: mask, slot init, h16 zero, sync init ----------------
__global__ void k_prep(const int* __restrict__ mask, const int* __restrict__ length,
                       float* __restrict__ m, int* __restrict__ s0,
                       unsigned* __restrict__ h16z, int* __restrict__ sync) {
  int idx = blockIdx.x * BLK + threadIdx.x;           // grid covers B*T
  if (idx < B * T) {
    int b = idx / T, t = idx % T;
    float v;
    if (t == length[b] - 1)      v = 1.0f;
    else if (t == 0)             v = 0.0f;
    else                         v = mask[idx] ? 1.0f : 0.0f;
    m[idx] = v;
    s0[idx] = -1;
  }
  if (idx < (2 * B * H) / 2) h16z[idx] = 0u;          // zero both h buffers (dwords)
  if (idx == 0) { sync[0] = 0; sync[1] = 0; }
}

// ---------------- gather-index build + word_nums ----------------
__global__ void k_gather(const float* __restrict__ m, int* __restrict__ s0,
                         float* __restrict__ dout, int K) {
  __shared__ float red[BLK];
  int b = blockIdx.x;
  float s = 0.f;
  for (int t = threadIdx.x; t < T; t += BLK) s += m[b * T + t];
  red[threadIdx.x] = s;
  __syncthreads();
  for (int st = BLK / 2; st > 0; st >>= 1) {
    if (threadIdx.x < st) red[threadIdx.x] += red[threadIdx.x + st];
    __syncthreads();
  }
  if (threadIdx.x == 0) {
    float wn = red[0];
    dout[(size_t)B * K * H + b] = wn;                 // word_nums tail
    int wni  = (int)(wn + 0.5f);
    int need = K - wni;                               // zeros to promote (last `need`)
    int Z    = T - wni;
    int zb = 0, k = 0;
    for (int t = 0; t < T && k < K; ++t) {
      float mv = m[b * T + t];
      bool sel;
      if (mv > 0.5f) sel = true;
      else { sel = (zb >= Z - need); ++zb; }
      if (sel) {
        int slot = b * K + k; ++k;
        int flat = b * T + t - 1;
        if (flat < 0) flat += B * T;                  // wrap, same as % (B*T)
        s0[flat] = slot;                              // proven unique per flat
      }
    }
  }
}

// ---------------- convert x to f16, (T,B,I) layout ----------------
__global__ void k_convx(const float* __restrict__ x, _Float16* __restrict__ x16) {
  size_t idx = (size_t)blockIdx.x * BLK + threadIdx.x;  // T*B*I
  if (idx < (size_t)T * B * I) {
    int i = (int)(idx % I);
    size_t rem = idx / I;
    int b = (int)(rem % B);
    int t = (int)(rem / B);
    x16[idx] = (_Float16)x[((size_t)b * T + t) * I + i];
  }
}

// ---------------- fuse + convert weights, sum biases ----------------
__global__ void k_convw(const float* __restrict__ wih, const float* __restrict__ whh,
                        const float* __restrict__ bih, const float* __restrict__ bhh,
                        _Float16* __restrict__ Wc, float* __restrict__ bsum) {
  int idx = blockIdx.x * BLK + threadIdx.x;            // G4*KD
  if (idx < G4 * KD) {
    int r = idx / KD, k = idx % KD;
    float v = (k < I) ? wih[r * I + k] : whh[r * H + (k - I)];
    Wc[idx] = (_Float16)v;
  }
  if (idx < G4) bsum[idx] = bih[idx] + bhh[idx];
}

// ---------------- grid-wide barrier (generation counter) ----------------
__device__ __forceinline__ void grid_barrier(int* sync) {
  __syncthreads();
  __builtin_amdgcn_fence(__ATOMIC_RELEASE, "agent");
  if (threadIdx.x == 0) {
    int* cnt = sync;
    int* gen = sync + 1;
    int g = __hip_atomic_load(gen, __ATOMIC_RELAXED, __HIP_MEMORY_SCOPE_AGENT);
    int prev = __hip_atomic_fetch_add(cnt, 1, __ATOMIC_ACQ_REL, __HIP_MEMORY_SCOPE_AGENT);
    if (prev == NWG - 1) {
      __hip_atomic_store(cnt, 0, __ATOMIC_RELAXED, __HIP_MEMORY_SCOPE_AGENT);
      __hip_atomic_fetch_add(gen, 1, __ATOMIC_RELEASE, __HIP_MEMORY_SCOPE_AGENT);
    } else {
      while (__hip_atomic_load(gen, __ATOMIC_ACQUIRE, __HIP_MEMORY_SCOPE_AGENT) == g)
        __builtin_amdgcn_s_sleep(2);
    }
  }
  __syncthreads();
  __builtin_amdgcn_fence(__ATOMIC_ACQUIRE, "agent");
}

// ---------------- persistent fused LSTM ----------------
// WG wg owns H-columns [wg*8, wg*8+8) -> 32 gate columns {g*H + wg*8 + j}.
// Wave w owns batch rows [w*16, w*16+16). Per step: C(16x32) = A(16x640) * Wslab^T
// via 20 K-chunks x 2 N-tiles of v_wmma_f32_16x16x32_f16. c-state lives in VGPRs.
constexpr int WROW = 660;  // padded LDS row (halves): 330 dwords, conflict-free B reads

__global__ void __launch_bounds__(BLK)
k_lstm(const _Float16* __restrict__ x16, _Float16* __restrict__ h16,
       const _Float16* __restrict__ Wc, const float* __restrict__ bsum,
       const float* __restrict__ mbuf, const int* __restrict__ slot0,
       const int* __restrict__ length, float* __restrict__ dout, int* sync) {
  __shared__ _Float16 wlds[32][WROW];

  const int tid  = threadIdx.x;
  const int wg   = blockIdx.x;       // 0..63
  const int wave = tid >> 5;         // 0..7 -> batch block
  const int lane = tid & 31;
  const int hi   = lane >> 4;        // lane half (K offset select)
  const int lr   = lane & 15;        // B/C column within tile

  // ---- stage this WG's 32 gate-rows of fused weights into LDS (once) ----
  for (int d = tid; d < 32 * (KD / 2); d += BLK) {
    int lc = d / (KD / 2), kd = d % (KD / 2);
    int grow = (lc >> 3) * H + wg * 8 + (lc & 7);     // gate*H + hcol
    ((unsigned*)&wlds[lc][0])[kd] = ((const unsigned*)(Wc + (size_t)grow * KD))[kd];
  }
  __syncthreads();

  const int   hcol  = wg * 8 + (lr & 7);
  const float bias0 = bsum[((lr < 8) ? 0 : 1) * H + hcol];  // i or f column
  const float bias1 = bsum[((lr < 8) ? 2 : 3) * H + hcol];  // g or o column
  const int   rbase = (lr < 8) ? 0 : 4;                     // acc rows this lane owns
  const int   browA = wave * 16 + lr;                       // A-fragment row

  int   len4[4], brow4[4];
  float creg[4] = {0.f, 0.f, 0.f, 0.f};
  float hreg[4] = {0.f, 0.f, 0.f, 0.f};
#pragma unroll
  for (int j = 0; j < 4; ++j) {
    int b = wave * 16 + hi * 8 + rbase + j;
    brow4[j] = b;
    len4[j]  = length[b];
  }

  int parity = 0;
  for (int t = 0; t < T; ++t) {
    v8f acc0 = {}; v8f acc1 = {};
    const _Float16* aX = x16 + ((size_t)t * B + browA) * I;
    const _Float16* aH = h16 + (size_t)parity * B * H + (size_t)browA * H;

    // Pull next timestep's x slab into WGP cache while this step's WMMAs run
    // (16 lanes x 256B-apart rows cover the wave's whole next-t A slab).
    if (t + 1 < T)
      __builtin_prefetch(x16 + ((size_t)(t + 1) * B + browA) * I, 0, 3);

#pragma unroll
    for (int kc = 0; kc < KD / 32; ++kc) {            // 20 chunks of K=32
      // A fragment: contiguous K order (same as B), K = 2v + (hi ? 16:0)
      F16x16 af;
      const _Float16* ap = (kc < 4) ? (aX + kc * 32) : (aH + (kc * 32 - I));
#pragma unroll
      for (int v = 0; v < 8; ++v) {
        int kk = 2 * v + (hi ? 16 : 0);
        af.u[v] = *(const unsigned*)(ap + kk);
      }
      // B fragments from LDS: lane column lr, K = 2v + (hi ? 16:0)
      F16x16 bf0, bf1;
      const unsigned* w0 = (const unsigned*)&wlds[lr][0];
      const unsigned* w1 = (const unsigned*)&wlds[16 + lr][0];
      int kbase = kc * 16 + (hi ? 8 : 0);
#pragma unroll
      for (int v = 0; v < 8; ++v) {
        bf0.u[v] = w0[kbase + v];
        bf1.u[v] = w1[kbase + v];
      }
      acc0 = __builtin_amdgcn_wmma_f32_16x16x32_f16(false, af.h, false, bf0.h,
                                                    (short)0, acc0, false, false);
      acc1 = __builtin_amdgcn_wmma_f32_16x16x32_f16(false, af.h, false, bf1.h,
                                                    (short)0, acc1, false, false);
    }

    // ---- gates -> state update (lane pairs L, L+8 exchange i/f and g/o) ----
    _Float16* hOut = h16 + (size_t)(parity ^ 1) * B * H;
#pragma unroll
    for (int r = 0; r < 8; ++r) {
      float x0 = acc0[r] + bias0;       // lr<8: i, lr>=8: f
      float x1 = acc1[r] + bias1;       // lr<8: g, lr>=8: o
      float p0 = __shfl_xor(x0, 8, 32);
      float p1 = __shfl_xor(x1, 8, 32);
      if ((r < 4) == (lr < 8)) {        // low lanes own rows 0-3, high rows 4-7
        int   j  = r & 3;
        int   b  = brow4[j];
        float iv = (lr < 8) ? x0 : p0;
        float fv = (lr < 8) ? p0 : x0;
        float gv = (lr < 8) ? x1 : p1;
        float ov = (lr < 8) ? p1 : x1;
        float mt = mbuf[(size_t)b * T + t];
        float cm = creg[j] * mt;
        float hm = hreg[j] * mt;
        float c2 = sigf(fv) * cm + sigf(iv) * tanhf(gv);
        float h2 = sigf(ov) * tanhf(c2);
        bool  act = (t < len4[j]);
        float hn = act ? h2 : hm;
        float cn = act ? c2 : cm;
        float e  = act ? h2 : 0.0f;
        creg[j] = cn;
        hreg[j] = hn;
        float mn = (t + 1 < T) ? mbuf[(size_t)b * T + t + 1] : 0.0f;
        hOut[(size_t)b * H + hcol] = (_Float16)(hn * mn);  // pre-masked for t+1
        int s0 = slot0[(size_t)b * T + t];
        if (s0 >= 0) dout[(size_t)s0 * H + hcol] = e;      // scatter emb row
      }
    }

    grid_barrier(sync);
    parity ^= 1;
  }
}

// ---------------- host launcher ----------------
extern "C" void kernel_launch(void* const* d_in, const int* in_sizes, int n_in,
                              void* d_out, int out_size, void* d_ws, size_t ws_size,
                              hipStream_t stream) {
  const float* x      = (const float*)d_in[0];
  const int*   mask   = (const int*)d_in[1];
  const int*   length = (const int*)d_in[2];
  const float* wih    = (const float*)d_in[3];
  const float* whh    = (const float*)d_in[4];
  const float* bih    = (const float*)d_in[5];
  const float* bhh    = (const float*)d_in[6];
  float*       out    = (float*)d_out;

  char* ws = (char*)d_ws;
  float*     m_f  = (float*)(ws + OFF_M);
  int*       s0   = (int*)(ws + OFF_S0);
  float*     bsum = (float*)(ws + OFF_BS);
  int*       sync = (int*)(ws + OFF_SY);
  _Float16*  h16  = (_Float16*)(ws + OFF_H16);
  _Float16*  Wc   = (_Float16*)(ws + OFF_WC);
  _Float16*  x16  = (_Float16*)(ws + OFF_X16);

  const int K = (out_size - B) / (B * H);   // recover K = max word count

  k_prep<<<(B * T + BLK - 1) / BLK, BLK, 0, stream>>>(mask, length, m_f, s0,
                                                      (unsigned*)h16, sync);
  k_convx<<<(int)(((size_t)T * B * I + BLK - 1) / BLK), BLK, 0, stream>>>(x, x16);
  k_convw<<<(G4 * KD + BLK - 1) / BLK, BLK, 0, stream>>>(wih, whh, bih, bhh, Wc, bsum);
  k_gather<<<B, BLK, 0, stream>>>(m_f, s0, out, K);
  k_lstm<<<NWG, BLK, 0, stream>>>(x16, h16, Wc, bsum, m_f, s0, length, out, sync);
}
